// SequencePredictorRecurrentTransformer_33174327394419
// MI455X (gfx1250) — matrix operand order repaired and checked by reading, not compiled, not run on hardware
//
#include <hip/hip_runtime.h>

// ---------------- model dims ----------------
#define TT   2048
#define BB   32
#define TB   (TT*BB)      // 65536 rows
#define DM   512
#define NH   8
#define DH   64
#define DFF  2048
#define NCLS 64
#define EV   256
#define CCH  128          // scan chunk length
#define NCH  (TT/CCH)     // 16 chunks
#define BHP  (BB*NH)      // 256 (b,h) pairs

typedef __bf16 bf16_t;
typedef bf16_t bf16x8  __attribute__((ext_vector_type(8)));
typedef bf16_t bf16x16 __attribute__((ext_vector_type(16)));
typedef float  f32x8   __attribute__((ext_vector_type(8)));
typedef unsigned int u32x4 __attribute__((ext_vector_type(4)));
typedef int    i32x4   __attribute__((ext_vector_type(4)));
typedef int    i32x8   __attribute__((ext_vector_type(8)));

// ---------------- workspace layout (bytes) ----------------
constexpr size_t SZ_W_DM  = (size_t)2*DM*DM*2;      // 2 layers, dm x dm, bf16
constexpr size_t SZ_W_FF  = (size_t)2*DM*DFF*2;
constexpr size_t OFF_WQ = 0;
constexpr size_t OFF_WK = OFF_WQ + SZ_W_DM;
constexpr size_t OFF_WV = OFF_WK + SZ_W_DM;
constexpr size_t OFF_WO = OFF_WV + SZ_W_DM;
constexpr size_t OFF_W1 = OFF_WO + SZ_W_DM;
constexpr size_t OFF_W2 = OFF_W1 + SZ_W_FF;
constexpr size_t OFF_WP = OFF_W2 + SZ_W_FF;
constexpr size_t OFF_H32 = (size_t)16<<20;
constexpr size_t SZ_ABF = (size_t)TB*DM*2;          // 64 MiB
constexpr size_t SZ_AF32= (size_t)TB*DM*4;          // 128 MiB
constexpr size_t OFF_HBF = OFF_H32 + SZ_AF32;
constexpr size_t OFF_Q   = OFF_HBF + SZ_ABF;
constexpr size_t OFF_K   = OFF_Q + SZ_ABF;
constexpr size_t OFF_V   = OFF_K + SZ_ABF;
constexpr size_t OFF_O   = OFF_V + SZ_ABF;
constexpr size_t OFF_FF  = OFF_Q;                   // alias q/k/v/o region (256 MiB)
constexpr size_t OFF_R32 = OFF_O + SZ_ABF;
constexpr size_t OFF_G   = OFF_R32 + SZ_AF32;       // per-chunk V^T K sums, f32
constexpr size_t OFF_ZG  = OFF_G  + (size_t)BHP*NCH*DH*DH*4;
constexpr size_t OFF_SP  = OFF_ZG + (size_t)BHP*NCH*DH*4;   // bf16 exclusive prefixes

// ---------------- bf16 helpers ----------------
__device__ inline float bf2f(bf16_t x){
  unsigned short u; __builtin_memcpy(&u,&x,2);
  unsigned int t=(unsigned)u<<16; float f; __builtin_memcpy(&f,&t,4); return f;
}
__device__ inline bf16_t f2bf(float f){
  unsigned int t; __builtin_memcpy(&t,&f,4);
  t += 0x7FFFu + ((t>>16)&1u);
  unsigned short u=(unsigned short)(t>>16);
  bf16_t r; __builtin_memcpy(&r,&u,2); return r;
}
__device__ inline f32x8 wmma_bf16(bf16x16 a, bf16x16 b, f32x8 c){
  return __builtin_amdgcn_wmma_f32_16x16x32_bf16(false, a, false, b, (short)0, c, false, false);
}

// ---------------- WMMA fragment loaders (ISA 7.12.2 lane maps) ----------------
// A 16x32 bf16: lane l -> row (l&15), K = k0 + (l>>4)*8 + {0..7, 16..23}
__device__ inline bf16x16 frag_A_g(const bf16_t* base, size_t ld, int row0, int k0, int lane){
  const bf16_t* p = base + (size_t)(row0 + (lane&15))*ld + k0 + ((lane>>4)<<3);
  union { bf16x16 v; bf16x8 h[2]; } u;
  u.h[0] = *(const bf16x8*)p;
  u.h[1] = *(const bf16x8*)(p + 16);
  return u.v;
}
// B 32x16 bf16 from row-major [N][K]: lane l -> col (l&15), K = k0 + (l>>4)*8 + {0..7, 16..23}
__device__ inline bf16x16 frag_B_g(const bf16_t* base, size_t ld, int n0, int k0, int lane){
  const bf16_t* p = base + (size_t)(n0 + (lane&15))*ld + k0 + ((lane>>4)<<3);
  union { bf16x16 v; bf16x8 h[2]; } u;
  u.h[0] = *(const bf16x8*)p;
  u.h[1] = *(const bf16x8*)(p + 16);
  return u.v;
}
__device__ inline bf16x16 frag_A_l(const char* s, int strideB, int row0, int k0, int lane){
  const char* p = s + (row0 + (lane&15))*strideB + (k0 + ((lane>>4)<<3))*2;
  union { bf16x16 v; bf16x8 h[2]; } u;
  u.h[0] = *(const bf16x8*)p;
  u.h[1] = *(const bf16x8*)(p + 32);
  return u.v;
}
__device__ inline bf16x16 frag_B_l(const char* s, int strideB, int n0, int k0, int lane){
  const char* p = s + (n0 + (lane&15))*strideB + (k0 + ((lane>>4)<<3))*2;
  union { bf16x16 v; bf16x8 h[2]; } u;
  u.h[0] = *(const bf16x8*)p;
  u.h[1] = *(const bf16x8*)(p + 32);
  return u.v;
}

// ---------------- TDM panel load: 32 rows x 256 bf16, LDS rows padded ----------------
// pad_interval = 128 dwords (512 B = one row), pad_amount = 4 dwords -> stride 528 B
constexpr int PANW = 256;            // K-panel width in elements
constexpr int SROW = 528;            // padded LDS row stride (33*16, conflict-free)
constexpr int PSZ  = 32*SROW;        // 16896 B per buffer

__device__ inline void tdm_panel(const bf16_t* A, int K, int m0, int p, unsigned lds){
  unsigned long long ga = (unsigned long long)(size_t)(A + (size_t)m0*K + (size_t)p*PANW);
  u32x4 g0;
  g0.x = 1u;                                   // count = 1 descriptor
  g0.y = lds;                                  // LDS byte address
  g0.z = (unsigned)(ga & 0xFFFFFFFFull);
  g0.w = (unsigned)((ga >> 32) & 0x01FFFFFFull) | (2u << 30);   // addr[56:32] | type=2
  i32x8 g1;
  g1[0] = (1<<16) | (1<<20) | (6<<22) | (3<<25); // data_size=2B, pad_en, 128dw ivl, 4dw pad
  g1[1] = (K & 0xFFFF) << 16;                    // tensor_dim0 lo
  g1[2] = (K >> 16);                             // tensor_dim0 hi | tensor_dim1 lo(=0)
  g1[3] = (TB >> 16) | (PANW << 16);             // tensor_dim1 hi | tile_dim0
  g1[4] = 32;                                    // tile_dim1 = 32 rows
  g1[5] = K;                                     // tensor_dim0_stride lo
  g1[6] = 0; g1[7] = 0;
  i32x4 z4 = {0,0,0,0};
#if defined(__clang_major__) && (__clang_major__ >= 23)
  i32x8 z8 = {0,0,0,0,0,0,0,0};
  __builtin_amdgcn_tensor_load_to_lds(g0, g1, z4, z4, z8, 0);
#else
  __builtin_amdgcn_tensor_load_to_lds(g0, g1, z4, z4, 0);
#endif
}

// ---------------- weight transpose + bf16 convert: Wt[n][k] = W[k][n] ----------------
__global__ void k_tcvt(const float* __restrict__ W, bf16_t* __restrict__ Wt, int K, int N){
  size_t idx = (size_t)blockIdx.x*256 + threadIdx.x;
  if (idx >= (size_t)K*N) return;
  int k = (int)(idx / N), n = (int)(idx % N);
  Wt[(size_t)n*K + k] = f2bf(W[idx]);
}

// ---------------- embedding + positional encoding ----------------
__global__ void k_embed(const int* __restrict__ x, const float* __restrict__ E,
                        float* __restrict__ h32, bf16_t* __restrict__ hbf){
  size_t idx = (size_t)blockIdx.x*256 + threadIdx.x;  // TB*DM elements
  int d = (int)(idx & (DM-1));
  size_t row = idx >> 9;
  int t = (int)(row >> 5), b = (int)(row & 31);
  float val;
  if (d < EV) {
    val = E[(size_t)x[t*BB + b]*EV + d];
  } else {
    int j = d - EV;
    int i2 = j >> 1;
    float div = __expf(-(float)(2*i2) * (9.2103403719761836f/256.f)); // ln(1e4)/256
    float ang = (float)t * div;
    val = (j & 1) ? __cosf(ang) : __sinf(ang);
  }
  h32[idx] = val;
  hbf[idx] = f2bf(val);
}

// ---------------- WMMA GEMM: C[M=TB, N] = act(A[TB,K] @ Bt[N,K]^T + bias) ----------
// Double-buffered TDM K-panels (DMA of panel p+1 overlaps WMMA on panel p).
// ACT: 0 none, 1 phi (elu+1), 2 relu.  OUTM: 0 f32, 1 bf16, 2 f32 permuted [B,T,N].
// NT: N-tiles (of 16) per wave; block covers 32 rows x 64*NT cols.
template<int ACT, int OUTM, int NT>
__global__ __launch_bounds__(128)
void k_gemm(const bf16_t* __restrict__ A, const bf16_t* __restrict__ Bt,
            const float* __restrict__ bias, float* __restrict__ Cf,
            bf16_t* __restrict__ Cbf, int K, int N){
  extern __shared__ char sA[];                 // 2 * PSZ bytes
  const int tid = threadIdx.x, lane = tid & 31, w = tid >> 5;
  const int m0 = blockIdx.x * 32;
  const int n0 = blockIdx.y * (64*NT) + w * (16*NT);
  const unsigned ldsb = (unsigned)(size_t)(void*)sA;
  f32x8 acc[2][NT];
  #pragma unroll
  for (int mt = 0; mt < 2; ++mt)
    #pragma unroll
    for (int nt = 0; nt < NT; ++nt) acc[mt][nt] = (f32x8){};
  const int npan = K / PANW;
  if (w == 0) tdm_panel(A, K, m0, 0, ldsb);
  for (int p = 0; p < npan; ++p) {
    const char* buf = sA + (p & 1)*PSZ;
    if (w == 0) {
      if (p + 1 < npan) {                      // kick DMA for next panel, then wait
        tdm_panel(A, K, m0, p + 1, ldsb + ((p + 1) & 1)*PSZ);
        __builtin_amdgcn_s_wait_tensorcnt(1);  // in-order TDM: panel p resident
      } else {
        __builtin_amdgcn_s_wait_tensorcnt(0);
      }
    }
    __syncthreads();
    #pragma unroll 4
    for (int kk = 0; kk < PANW; kk += 32) {
      bf16x16 a0 = frag_A_l(buf, SROW, 0,  kk, lane);
      bf16x16 a1 = frag_A_l(buf, SROW, 16, kk, lane);
      #pragma unroll
      for (int nt = 0; nt < NT; ++nt) {
        bf16x16 b = frag_B_g(Bt, (size_t)K, n0 + nt*16, p*PANW + kk, lane);
        acc[0][nt] = wmma_bf16(a0, b, acc[0][nt]);
        acc[1][nt] = wmma_bf16(a1, b, acc[1][nt]);
      }
    }
    __syncthreads();                           // buffer free for DMA of p+2
  }
  #pragma unroll
  for (int nt = 0; nt < NT; ++nt) {
    const int ncol = n0 + nt*16 + (lane & 15);
    const float bv = bias[ncol];
    #pragma unroll
    for (int r = 0; r < 8; ++r) {
      int row0 = m0 + r + ((lane>>4)<<3);
      int row1 = row0 + 16;
      float v0 = acc[0][nt][r] + bv, v1 = acc[1][nt][r] + bv;
      if (ACT == 1) { v0 = v0 > 0.f ? v0 + 1.f : __expf(v0);
                      v1 = v1 > 0.f ? v1 + 1.f : __expf(v1); }
      if (ACT == 2) { v0 = fmaxf(v0, 0.f); v1 = fmaxf(v1, 0.f); }
      if (OUTM == 0) {
        Cf[(size_t)row0*N + ncol] = v0;
        Cf[(size_t)row1*N + ncol] = v1;
      } else if (OUTM == 1) {
        Cbf[(size_t)row0*N + ncol] = f2bf(v0);
        Cbf[(size_t)row1*N + ncol] = f2bf(v1);
      } else { // permuted projection store: row = t*B+b -> out[b][t][n]
        Cf[((size_t)(row0 & 31)*TT + (row0 >> 5))*NCLS + ncol] = v0;
        Cf[((size_t)(row1 & 31)*TT + (row1 >> 5))*NCLS + ncol] = v1;
      }
    }
  }
}

// ---------------- scan phase A: per-chunk G = V^T K via WMMA, Zg = colsum(K) --------
__global__ __launch_bounds__(256)
void k_scanA(const bf16_t* __restrict__ kk, const bf16_t* __restrict__ v,
             float* __restrict__ G, float* __restrict__ Zg){
  extern __shared__ char sm[];
  char* skT = sm;                            // [64][272 B]  K^T: row d, col t
  char* svT = sm + 64*272;                   // [64][272 B]  V^T: row m, col t
  const int tid = threadIdx.x, lane = tid & 31, w = tid >> 5;
  const int c = blockIdx.x, bh = blockIdx.y;
  const int b = bh >> 3, h = bh & 7;
  const size_t rs = (size_t)BB*DM;
  const size_t base = ((size_t)(c*CCH)*BB + b)*DM + (size_t)h*DH;
  const bf16_t* kb = kk + base;
  const bf16_t* vb = v  + base;
  for (int idx = tid; idx < CCH*DH; idx += 256) {
    int t = idx >> 6, d = idx & 63;
    *(bf16_t*)(skT + d*272 + t*2) = kb[(size_t)t*rs + d];
    *(bf16_t*)(svT + d*272 + t*2) = vb[(size_t)t*rs + d];
  }
  __syncthreads();
  float* Gc = G + ((size_t)bh*NCH + c)*DH*DH;
  // 16 output tiles (4x4 of 16x16); wave w computes 2 of them with K-dim = t (128)
  #pragma unroll
  for (int u = 0; u < 2; ++u) {
    const int ti = w*2 + u;
    const int mi = ti >> 2, ji = ti & 3;
    f32x8 acc = {};
    #pragma unroll
    for (int ks = 0; ks < 4; ++ks) {
      bf16x16 a  = frag_A_l(svT, 272, mi*16, ks*32, lane);   // A rows = m of V^T
      bf16x16 bf = frag_B_l(skT, 272, ji*16, ks*32, lane);   // B cols = d of K
      acc = wmma_bf16(a, bf, acc);
    }
    const int d  = ji*16 + (lane & 15);
    const int mb = mi*16 + ((lane>>4)<<3);
    #pragma unroll
    for (int r = 0; r < 8; ++r)
      Gc[(size_t)(mb + r)*DH + d] = acc[r];
  }
  if (tid < DH) {                              // Z: column sums of K
    float acc = 0.f;
    const char* row = skT + tid*272;
    for (int t = 0; t < CCH; ++t) acc += bf2f(*(const bf16_t*)(row + t*2));
    Zg[((size_t)bh*NCH + c)*DH + tid] = acc;
  }
}

// ---------------- scan phase B: exclusive prefix over chunks ----------------
__global__ __launch_bounds__(256)
void k_scanB(const float* __restrict__ G, float* __restrict__ Zg,
             bf16_t* __restrict__ Spre){
  const int tid = threadIdx.x, bh = blockIdx.x;
  for (int r = 0; r < 16; ++r) {
    int e = tid + 256*r;
    float run = 0.f;
    for (int c = 0; c < NCH; ++c) {
      size_t o = ((size_t)bh*NCH + c)*DH*DH + e;
      float gv = G[o];
      Spre[o] = f2bf(run);     // ST layout: Spre[m][d] (row-major, contiguous d)
      run += gv;
    }
  }
  if (tid < DH) {
    float run = 0.f;
    for (int c = 0; c < NCH; ++c) {
      size_t o = ((size_t)bh*NCH + c)*DH + tid;
      float zv = Zg[o]; Zg[o] = run; run += zv;   // in place: element-owned
    }
  }
}

// ---------------- scan phase C: P = QK^T masked; O = (P V + Q S)/den ----------------
// den = eps + q.Z_pre + rowsum(P_mask); rowsum done as P_mask @ ones via WMMA,
// so every lane holds its own rows' denominator (no shuffles, no LDS round-trip).
__global__ __launch_bounds__(256)
void k_scanC(const bf16_t* __restrict__ q, const bf16_t* __restrict__ kk,
             const bf16_t* __restrict__ v, const bf16_t* __restrict__ spre,
             const float* __restrict__ zpre, bf16_t* __restrict__ o){
  extern __shared__ char sm[];
  char*  sP   = sm;                         // 128 x 272 B (bf16, padded rows)
  char*  svT  = sm + 128*272;               // 64 x 272 B
  float* sden = (float*)(sm + 128*272 + 64*272);  // [128] cross-term part
  const int tid = threadIdx.x, lane = tid & 31, w = tid >> 5;
  const int c = blockIdx.x, bh = blockIdx.y;
  const int b = bh >> 3, h = bh & 7;
  const size_t rs = (size_t)BB*DM;
  const size_t base = ((size_t)(c*CCH)*BB + b)*DM + (size_t)h*DH;
  const bf16_t* qb = q  + base;
  const bf16_t* kb = kk + base;
  const bf16_t* vb = v  + base;
  bf16_t*       ob = o  + base;
  // stage V transposed: svT[d][t]
  for (int idx = tid; idx < CCH*DH; idx += 256) {
    int t = idx >> 6, d = idx & 63;
    *(bf16_t*)(svT + d*272 + t*2) = vb[(size_t)t*rs + d];
  }
  // den cross part: eps + q . Z_pre
  const float* zp = zpre + ((size_t)bh*NCH + c)*DH;
  for (int t = tid; t < CCH; t += 256) {
    float acc = 1e-6f;
    for (int d = 0; d < DH; ++d) acc += bf2f(qb[(size_t)t*rs + d]) * zp[d];
    sden[t] = acc;
  }
  __syncthreads();
  // --- P phase: wave w owns row-tile i = w (rows 16i..16i+15) ---
  const int i = w;
  const int tb = i*16 + ((lane>>4)<<3);
  bf16x16 aq0 = frag_A_g(qb, rs, i*16, 0,  lane);
  bf16x16 aq1 = frag_A_g(qb, rs, i*16, 32, lane);
  for (int j = 0; j < 8; ++j) {
    f32x8 pc = {};
    bf16x16 bk0 = frag_B_g(kb, rs, j*16, 0,  lane);
    bf16x16 bk1 = frag_B_g(kb, rs, j*16, 32, lane);
    pc = wmma_bf16(aq0, bk0, pc);
    pc = wmma_bf16(aq1, bk1, pc);
    const int s = j*16 + (lane & 15);
    #pragma unroll
    for (int r = 0; r < 8; ++r) {
      float val = (s <= tb + r) ? pc[r] : 0.f;   // causal mask (inclusive)
      *(bf16_t*)(sP + (tb + r)*272 + s*2) = f2bf(val);
    }
  }
  __syncthreads();
  // --- den: rowsum(P_mask) = P_mask @ ones (D columns identical per row) ---
  const bf16_t one_ = f2bf(1.f);
  bf16x16 vones;
  #pragma unroll
  for (int e = 0; e < 16; ++e) vones[e] = one_;
  f32x8 ncd = {};
  #pragma unroll
  for (int ks = 0; ks < 4; ++ks) {
    bf16x16 ap = frag_A_l(sP, 272, i*16, ks*32, lane);
    ncd = wmma_bf16(ap, vones, ncd);
  }
  float vden[8];
  #pragma unroll
  for (int r = 0; r < 8; ++r) vden[r] = sden[tb + r] + ncd[r];
  // --- num phase: O tiles (i, j) for j in 0..3 ---
  const bf16_t* sp = spre + ((size_t)bh*NCH + c)*DH*DH;   // ST[m][d], ld=64
  for (int j = 0; j < 4; ++j) {
    f32x8 nc = {};
    bf16x16 bs0 = frag_B_g(sp, (size_t)DH, j*16, 0,  lane);
    bf16x16 bs1 = frag_B_g(sp, (size_t)DH, j*16, 32, lane);
    nc = wmma_bf16(aq0, bs0, nc);               // cross-chunk: Q @ S_pre
    nc = wmma_bf16(aq1, bs1, nc);
    #pragma unroll
    for (int ks = 0; ks < 4; ++ks) {            // intra-chunk: P_mask @ V
      bf16x16 ap = frag_A_l(sP,  272, i*16, ks*32, lane);
      bf16x16 bv = frag_B_l(svT, 272, j*16, ks*32, lane);
      nc = wmma_bf16(ap, bv, nc);
    }
    const int m = j*16 + (lane & 15);
    #pragma unroll
    for (int r = 0; r < 8; ++r)
      ob[(size_t)(tb + r)*rs + m] = f2bf(nc[r] / vden[r]);
  }
}

// ---------------- residual add + LayerNorm (one wave per row) ----------------
__global__ __launch_bounds__(256)
void k_addln(float* __restrict__ h32, bf16_t* __restrict__ hbf,
             const float* __restrict__ delta, const float* __restrict__ g,
             const float* __restrict__ be){
  const int tid = threadIdx.x, lane = tid & 31, w = tid >> 5;
  const size_t row = (size_t)blockIdx.x*8 + w;
  const float* hr = h32 + row*DM;
  const float* dr = delta + row*DM;
  float xv[16]; float s = 0.f, sq = 0.f;
  #pragma unroll
  for (int i = 0; i < 16; ++i) {
    float vv = hr[lane + i*32] + dr[lane + i*32];
    xv[i] = vv; s += vv; sq += vv*vv;
  }
  #pragma unroll
  for (int m = 1; m < 32; m <<= 1) { s += __shfl_xor(s, m); sq += __shfl_xor(sq, m); }
  float mean = s * (1.f/DM);
  float var  = sq * (1.f/DM) - mean*mean;
  float rstd = rsqrtf(var + 1e-5f);
  float* ho = h32 + row*DM; bf16_t* hb = hbf + row*DM;
  #pragma unroll
  for (int i = 0; i < 16; ++i) {
    int d = lane + i*32;
    float y = (xv[i] - mean)*rstd*g[d] + be[d];
    ho[d] = y; hb[d] = f2bf(y);
  }
}

// ---------------- launch ----------------
extern "C" void kernel_launch(void* const* d_in, const int* in_sizes, int n_in,
                              void* d_out, int out_size, void* d_ws, size_t ws_size,
                              hipStream_t stream){
  (void)in_sizes; (void)n_in; (void)out_size; (void)ws_size;
  const int*   x   = (const int*)  d_in[0];
  const float* E   = (const float*)d_in[1];
  const float* Wq  = (const float*)d_in[2];  const float* bq = (const float*)d_in[3];
  const float* Wk  = (const float*)d_in[4];  const float* bk = (const float*)d_in[5];
  const float* Wv  = (const float*)d_in[6];  const float* bv = (const float*)d_in[7];
  const float* Wo  = (const float*)d_in[8];  const float* bo = (const float*)d_in[9];
  const float* g1  = (const float*)d_in[10]; const float* be1= (const float*)d_in[11];
  const float* W1  = (const float*)d_in[12]; const float* b1 = (const float*)d_in[13];
  const float* W2  = (const float*)d_in[14]; const float* b2 = (const float*)d_in[15];
  const float* g2  = (const float*)d_in[16]; const float* be2= (const float*)d_in[17];
  const float* Wp  = (const float*)d_in[18]; const float* bp = (const float*)d_in[19];

  char* ws = (char*)d_ws;
  bf16_t* WqT = (bf16_t*)(ws + OFF_WQ);
  bf16_t* WkT = (bf16_t*)(ws + OFF_WK);
  bf16_t* WvT = (bf16_t*)(ws + OFF_WV);
  bf16_t* WoT = (bf16_t*)(ws + OFF_WO);
  bf16_t* W1T = (bf16_t*)(ws + OFF_W1);
  bf16_t* W2T = (bf16_t*)(ws + OFF_W2);
  bf16_t* WpT = (bf16_t*)(ws + OFF_WP);
  float*  h32 = (float*) (ws + OFF_H32);
  bf16_t* hbf = (bf16_t*)(ws + OFF_HBF);
  bf16_t* qbf = (bf16_t*)(ws + OFF_Q);
  bf16_t* kbf = (bf16_t*)(ws + OFF_K);
  bf16_t* vbf = (bf16_t*)(ws + OFF_V);
  bf16_t* obf = (bf16_t*)(ws + OFF_O);
  bf16_t* ffbf= (bf16_t*)(ws + OFF_FF);
  float*  r32 = (float*) (ws + OFF_R32);
  float*  G   = (float*) (ws + OFF_G);
  float*  Zg  = (float*) (ws + OFF_ZG);
  bf16_t* Spre= (bf16_t*)(ws + OFF_SP);

  auto TC = [&](const float* W, bf16_t* Wt, int K, int N){
    size_t tot = (size_t)K*N;
    k_tcvt<<<dim3((unsigned)((tot + 255)/256)), 256, 0, stream>>>(W, Wt, K, N);
  };
  for (int l = 0; l < 2; ++l) {
    TC(Wq + (size_t)l*DM*DM,  WqT + (size_t)l*DM*DM,  DM, DM);
    TC(Wk + (size_t)l*DM*DM,  WkT + (size_t)l*DM*DM,  DM, DM);
    TC(Wv + (size_t)l*DM*DM,  WvT + (size_t)l*DM*DM,  DM, DM);
    TC(Wo + (size_t)l*DM*DM,  WoT + (size_t)l*DM*DM,  DM, DM);
    TC(W1 + (size_t)l*DM*DFF, W1T + (size_t)l*DFF*DM, DM, DFF);
    TC(W2 + (size_t)l*DFF*DM, W2T + (size_t)l*DM*DFF, DFF, DM);
  }
  TC(Wp, WpT, DM, NCLS);

  k_embed<<<dim3((unsigned)((size_t)TB*DM/256)), 256, 0, stream>>>(x, E, h32, hbf);

  const size_t gemm_lds  = (size_t)2*PSZ;                 // double-buffered panels
  const size_t scanA_lds = (size_t)2*64*272;
  const size_t scanC_lds = (size_t)128*272 + 64*272 + CCH*4;

  for (int l = 0; l < 2; ++l) {
    const size_t wdm = (size_t)l*DM*DM, wff = (size_t)l*DFF*DM;
    k_gemm<1,1,2><<<dim3(TB/32, DM/128), 128, gemm_lds, stream>>>(hbf, WqT+wdm, bq+l*DM, nullptr, qbf, DM, DM);
    k_gemm<1,1,2><<<dim3(TB/32, DM/128), 128, gemm_lds, stream>>>(hbf, WkT+wdm, bk+l*DM, nullptr, kbf, DM, DM);
    k_gemm<0,1,2><<<dim3(TB/32, DM/128), 128, gemm_lds, stream>>>(hbf, WvT+wdm, bv+l*DM, nullptr, vbf, DM, DM);
    k_scanA<<<dim3(NCH, BHP), 256, scanA_lds, stream>>>(kbf, vbf, G, Zg);
    k_scanB<<<dim3(BHP), 256, 0, stream>>>(G, Zg, Spre);
    k_scanC<<<dim3(NCH, BHP), 256, scanC_lds, stream>>>(qbf, kbf, vbf, Spre, Zg, obf);
    k_gemm<0,0,2><<<dim3(TB/32, DM/128), 128, gemm_lds, stream>>>(obf, WoT+wdm, bo+l*DM, r32, nullptr, DM, DM);
    k_addln<<<dim3(TB/8), 256, 0, stream>>>(h32, hbf, r32, g1+l*DM, be1+l*DM);
    k_gemm<2,1,2><<<dim3(TB/32, DFF/128), 128, gemm_lds, stream>>>(hbf, W1T+wff, b1+l*DFF, nullptr, ffbf, DM, DFF);
    k_gemm<0,0,2><<<dim3(TB/32, DM/128), 128, gemm_lds, stream>>>(ffbf, W2T+wff, b2+l*DM, r32, nullptr, DFF, DM);
    k_addln<<<dim3(TB/8), 256, 0, stream>>>(h32, hbf, r32, g2+l*DM, be2+l*DM);
  }
  k_gemm<0,2,1><<<dim3(TB/32, 1), 128, gemm_lds, stream>>>(hbf, WpT, bp, (float*)d_out, nullptr, DM, NCLS);
}
